// MultiHeadAttentionLayer_35433480192360
// MI455X (gfx1250) — matrix operand order repaired and checked by reading, not compiled, not run on hardware
//
#include <hip/hip_runtime.h>

// ---------------------------------------------------------------- types ----
typedef __attribute__((ext_vector_type(16))) __bf16 v16bf;
typedef __attribute__((ext_vector_type(8)))  float  v8f;

union FragU {
    v16bf v;
    uint4 q[2];
};

// Model dims
#define BATCH   4
#define SEQ     2048
#define DMODEL  1024
#define NHEADS  16
#define HDIM    64
#define MROWS   (BATCH * SEQ)      // 8192
#define THREED  (3 * DMODEL)       // 3072

// Workspace layout (bytes)
#define XB_OFF     0ull
#define WQKVT_OFF  (XB_OFF    + (size_t)MROWS * DMODEL * 2)
#define WOUTT_OFF  (WQKVT_OFF + (size_t)THREED * DMODEL * 2)
#define Q_OFF      (WOUTT_OFF + (size_t)DMODEL * DMODEL * 2)
#define K_OFF      (Q_OFF     + (size_t)BATCH * NHEADS * SEQ * HDIM * 2)
#define VT_OFF     (K_OFF     + (size_t)BATCH * NHEADS * SEQ * HDIM * 2)
#define CTX_OFF    (VT_OFF    + (size_t)BATCH * NHEADS * SEQ * HDIM * 2)

// ------------------------------------------ async global->LDS (CDNA5) ------
#if defined(__has_builtin)
#  if __has_builtin(__builtin_amdgcn_global_load_async_to_lds_b128)
#    define HAVE_ASYNC_LDS 1
#  endif
#endif
#ifndef HAVE_ASYNC_LDS
#  define HAVE_ASYNC_LDS 0
#endif

#if HAVE_ASYNC_LDS
typedef int v4i_a __attribute__((vector_size(16)));
#  define ASYNC_COPY16(dst, src)                                             \
      __builtin_amdgcn_global_load_async_to_lds_b128(                        \
          (__attribute__((address_space(1))) v4i_a*)(src),                   \
          (__attribute__((address_space(3))) v4i_a*)(dst), 0, 0)
#  if __has_builtin(__builtin_amdgcn_s_wait_asynccnt)
#    define WAIT_ASYNC(n) __builtin_amdgcn_s_wait_asynccnt(n)
#  else
#    define WAIT_ASYNC(n) asm volatile("s_wait_asynccnt %0" ::"i"(n) : "memory")
#  endif
#else
#  define WAIT_ASYNC(n) do {} while (0)
#endif

// copy 32 halfwords (64B) global -> LDS for one thread
__device__ __forceinline__ void copy_row32(unsigned short* lds,
                                           const unsigned short* g) {
#if HAVE_ASYNC_LDS
    ASYNC_COPY16(lds, g);
    ASYNC_COPY16(lds + 8, g + 8);
    ASYNC_COPY16(lds + 16, g + 16);
    ASYNC_COPY16(lds + 24, g + 24);
#else
    const uint4* s = (const uint4*)g;
    uint4 r0 = s[0], r1 = s[1], r2 = s[2], r3 = s[3];
    uint4* d = (uint4*)lds;
    d[0] = r0; d[1] = r1; d[2] = r2; d[3] = r3;
#endif
}

// copy 16 halfwords (32B) global -> LDS for one thread
__device__ __forceinline__ void copy_row16(unsigned short* lds,
                                           const unsigned short* g) {
#if HAVE_ASYNC_LDS
    ASYNC_COPY16(lds, g);
    ASYNC_COPY16(lds + 8, g + 8);
#else
    const uint4* s = (const uint4*)g;
    uint4 r0 = s[0], r1 = s[1];
    uint4* d = (uint4*)lds;
    d[0] = r0; d[1] = r1;
#endif
}

// ------------------------------------------------------------- helpers ----
__device__ __forceinline__ unsigned short f2bf(float f) {
    unsigned int u = __float_as_uint(f);
    unsigned int r = u + 0x7fffu + ((u >> 16) & 1u);
    return (unsigned short)(r >> 16);
}

__device__ __forceinline__ v8f wmma_bf16(v16bf a, v16bf b, v8f c) {
    return __builtin_amdgcn_wmma_f32_16x16x32_bf16(
        false, a, false, b, (short)0, c, false, false);
}

// A fragment (16x32 bf16): lane l holds row m = row_base + (l&15);
// e<8 -> k = kbase + (l>>4)*8 + e ; e>=8 -> k = kbase + 16 + (l>>4)*8 + (e-8)
__device__ __forceinline__ v16bf load_frag_a(const unsigned short* lds, int stride,
                                             int row_base, int kbase, int lane) {
    int half = lane >> 4;
    const unsigned short* p = lds + (size_t)(row_base + (lane & 15)) * stride
                                  + kbase + half * 8;
    FragU f;
    f.q[0] = *(const uint4*)(p);
    f.q[1] = *(const uint4*)(p + 16);
    return f.v;
}

// B fragment ([n][k] LDS): lane l holds col n = n_base + (l&15);
// e<8 -> k = kbase + (l>>4)*8 + e ; e>=8 -> k = kbase + 16 + (l>>4)*8 + (e-8)
__device__ __forceinline__ v16bf load_frag_b(const unsigned short* lds, int stride,
                                             int n_base, int kbase, int lane) {
    int half = lane >> 4;
    const unsigned short* p = lds + (size_t)(n_base + (lane & 15)) * stride
                                  + kbase + half * 8;
    FragU f;
    f.q[0] = *(const uint4*)(p);
    f.q[1] = *(const uint4*)(p + 16);
    return f.v;
}

// --------------------------------------------------- conversion kernels ----
__global__ __launch_bounds__(256) void convert_f32_bf16_kernel(
    const float* __restrict__ x, unsigned short* __restrict__ xb, int n) {
    int i = (blockIdx.x * 256 + threadIdx.x) * 4;
    if (i + 3 < n) {
        float4 v = *(const float4*)(x + i);
        ushort4 o;
        o.x = f2bf(v.x); o.y = f2bf(v.y); o.z = f2bf(v.z); o.w = f2bf(v.w);
        *(ushort4*)(xb + i) = o;
    }
}

// W [K,N] f32 row-major  ->  Wt [N,K] bf16 row-major
__global__ __launch_bounds__(256) void transpose_f32_bf16_kernel(
    const float* __restrict__ W, unsigned short* __restrict__ Wt, int K, int N) {
    __shared__ unsigned short tile[32][33];
    int n0 = blockIdx.x * 32, k0 = blockIdx.y * 32;
    int tx = threadIdx.x, ty = threadIdx.y;  // 32 x 8
#pragma unroll
    for (int i = 0; i < 32; i += 8)
        tile[ty + i][tx] = f2bf(W[(size_t)(k0 + ty + i) * N + n0 + tx]);
    __syncthreads();
#pragma unroll
    for (int i = 0; i < 32; i += 8)
        Wt[(size_t)(n0 + ty + i) * K + k0 + tx] = tile[tx][ty + i];
}

// --------------------------------------------------------- WMMA GEMM ------
// C[M,N] = A[M,K](bf16) * Bt[N,K]^T(bf16) + bias
// 128x128 block tile, BK=64, double-buffered LDS + async copies.
// MODE 0: QKV scatter epilogue (bf16 q/k/vt) ; MODE 1: fp32 epilogue
template <int MODE>
__global__ __launch_bounds__(256) void gemm_bf16_kernel(
    const unsigned short* __restrict__ A, const unsigned short* __restrict__ Bt,
    const float* __restrict__ bias, float* __restrict__ outF,
    unsigned short* __restrict__ qOut, unsigned short* __restrict__ kOut,
    unsigned short* __restrict__ vtOut, int M, int N, int K) {
    constexpr int ST = 72;  // padded row stride (halfwords) for BK=64
    __shared__ __align__(16) unsigned short As[2][128 * ST];
    __shared__ __align__(16) unsigned short Bs[2][128 * ST];

    const int tid = threadIdx.x;
    const int lane = tid & 31;
    const int wave = tid >> 5;
    const int half = lane >> 4;
    const int l15 = lane & 15;
    const int m0 = blockIdx.y * 128;
    const int n0 = blockIdx.x * 128;
    const int wm = wave >> 2;  // 0..1 -> 64 rows
    const int wn = wave & 3;   // 0..3 -> 32 cols

    v8f acc[4][2];
#pragma unroll
    for (int i = 0; i < 4; ++i)
#pragma unroll
        for (int j = 0; j < 2; ++j) acc[i][j] = (v8f)(0.0f);

    // tile staging: 128 rows, 2 threads/row, 32 halfwords each
    const int lrow = tid >> 1;
    const int lk = (tid & 1) * 32;
    const unsigned short* gA = A + (size_t)(m0 + lrow) * K + lk;
    const unsigned short* gB = Bt + (size_t)(n0 + lrow) * K + lk;
    const int NT = K / 64;

    copy_row32(&As[0][lrow * ST + lk], gA);
    copy_row32(&Bs[0][lrow * ST + lk], gB);

    for (int t = 0; t < NT; ++t) {
        const int cur = t & 1;
        if (t + 1 < NT) {
            copy_row32(&As[cur ^ 1][lrow * ST + lk], gA + (t + 1) * 64);
            copy_row32(&Bs[cur ^ 1][lrow * ST + lk], gB + (t + 1) * 64);
            WAIT_ASYNC(8);  // tile t's 8 async copies done (in-order)
        } else {
            WAIT_ASYNC(0);
        }
        __syncthreads();

#pragma unroll
        for (int kk = 0; kk < 64; kk += 32) {
            v16bf af[4], bfr[2];
#pragma unroll
            for (int i = 0; i < 4; ++i)
                af[i] = load_frag_a(&As[cur][0], ST, wm * 64 + i * 16, kk, lane);
#pragma unroll
            for (int j = 0; j < 2; ++j)
                bfr[j] = load_frag_b(&Bs[cur][0], ST, wn * 32 + j * 16, kk, lane);
#pragma unroll
            for (int i = 0; i < 4; ++i)
#pragma unroll
                for (int j = 0; j < 2; ++j)
                    acc[i][j] = wmma_bf16(af[i], bfr[j], acc[i][j]);
        }
        __syncthreads();  // done reading buf cur -> safe for tile t+2 copies
    }

    // epilogue
#pragma unroll
    for (int i = 0; i < 4; ++i) {
#pragma unroll
        for (int j = 0; j < 2; ++j) {
            const int gn = n0 + wn * 32 + j * 16 + l15;
            const float bv = bias[gn];
#pragma unroll
            for (int r = 0; r < 8; ++r) {
                const int gm = m0 + wm * 64 + i * 16 + r + 8 * half;
                const float v = acc[i][j][r] + bv;
                if (MODE == 0) {
                    const int h = gn / 192;
                    const int rem = gn - h * 192;
                    const int which = rem >> 6;
                    const int d = rem & 63;
                    const int b = gm >> 11;
                    const int s = gm & (SEQ - 1);
                    const size_t bh = (size_t)b * NHEADS + h;
                    const unsigned short o = f2bf(v);
                    if (which == 0)
                        qOut[(bh * SEQ + s) * HDIM + d] = o;
                    else if (which == 1)
                        kOut[(bh * SEQ + s) * HDIM + d] = o;
                    else
                        vtOut[(bh * HDIM + d) * SEQ + s] = o;
                } else {
                    outF[(size_t)gm * N + gn] = v;
                }
            }
        }
    }
}

// ------------------------------------------------ flash attention kernel ---
// grid: (SEQ/64, BATCH*NHEADS), block: 256
__global__ __launch_bounds__(256) void attn_kernel(
    const unsigned short* __restrict__ q, const unsigned short* __restrict__ k,
    const unsigned short* __restrict__ vt, unsigned short* __restrict__ ctx) {
    constexpr int QS = 72, KS = 72, VS = 72, PS = 72;  // halfword strides
    constexpr int SS = 68;                              // fp32 stride
    __shared__ __align__(16) unsigned short Qs[64 * QS];
    __shared__ __align__(16) unsigned short Ks[2][64 * KS];
    __shared__ __align__(16) unsigned short Vts[2][64 * VS];
    __shared__ __align__(16) unsigned short Ps[64 * PS];
    __shared__ float Ss[64 * SS];
    __shared__ float mrow[64], lrow[64], arow[64];
    __shared__ float pmax[4][64], psum[4][64];

    const int tid = threadIdx.x;
    const int lane = tid & 31;
    const int wave = tid >> 5;
    const int half = lane >> 4;
    const int l15 = lane & 15;

    const int qt = blockIdx.x;
    const int bh = blockIdx.y;
    const int b = bh >> 4;
    const int h = bh & 15;

    const unsigned short* qp = q + (size_t)bh * SEQ * HDIM + (size_t)qt * 64 * HDIM;
    const unsigned short* kp = k + (size_t)bh * SEQ * HDIM;
    const unsigned short* vp = vt + (size_t)bh * HDIM * SEQ;

    // tile staging: 64 rows, 4 threads/row, 16 halfwords each
    const int row4 = tid >> 2;
    const int c16 = (tid & 3) * 16;

    // issue K/V tile 0 (async path overlaps with Q load + frag setup)
    copy_row16(&Ks[0][row4 * KS + c16], kp + (size_t)row4 * HDIM + c16);
    copy_row16(&Vts[0][row4 * VS + c16], vp + (size_t)row4 * SEQ + c16);

    // load Q tile 64x64 (synchronous; consumed immediately as register frags)
    {
        const uint4* g = (const uint4*)(qp + row4 * HDIM + c16);
        uint4 a0 = g[0], a1 = g[1];
        *(uint4*)&Qs[row4 * QS + c16] = a0;
        *(uint4*)&Qs[row4 * QS + c16 + 8] = a1;
    }
    if (tid < 64) { mrow[tid] = -3.0e38f; lrow[tid] = 0.0f; }
    __syncthreads();

    const int fi = wave >> 1;        // m-frag (query rows fi*16..)
    const int fj0 = (wave & 1) * 2;  // n-frag base

    v16bf qa0 = load_frag_a(Qs, QS, fi * 16, 0, lane);
    v16bf qa1 = load_frag_a(Qs, QS, fi * 16, 32, lane);

    v8f accO[2];
    accO[0] = (v8f)(0.0f);
    accO[1] = (v8f)(0.0f);

    constexpr int NT = SEQ / 64;  // 32
    for (int it = 0; it < NT; ++it) {
        const int cur = it & 1;
        if (it + 1 < NT) {
            const int kvn = (it + 1) * 64;
            copy_row16(&Ks[cur ^ 1][row4 * KS + c16],
                       kp + (size_t)(kvn + row4) * HDIM + c16);
            copy_row16(&Vts[cur ^ 1][row4 * VS + c16],
                       vp + (size_t)row4 * SEQ + kvn + c16);
            WAIT_ASYNC(4);  // tile it's 4 async copies complete (in-order)
        } else {
            WAIT_ASYNC(0);
        }
        __syncthreads();

        // ---- scores: 64x64, each wave owns 2 (16x16) fragments
        v8f sc[2];
        sc[0] = (v8f)(0.0f);
        sc[1] = (v8f)(0.0f);
#pragma unroll
        for (int jj = 0; jj < 2; ++jj) {
            const int nf = fj0 + jj;
            v16bf b0 = load_frag_b(&Ks[cur][0], KS, nf * 16, 0, lane);
            v16bf b1 = load_frag_b(&Ks[cur][0], KS, nf * 16, 32, lane);
            sc[jj] = wmma_bf16(qa0, b0, sc[jj]);
            sc[jj] = wmma_bf16(qa1, b1, sc[jj]);
        }
#pragma unroll
        for (int jj = 0; jj < 2; ++jj) {
            const int col = (fj0 + jj) * 16 + l15;
#pragma unroll
            for (int r = 0; r < 8; ++r) {
                const int row = fi * 16 + r + 8 * half;
                Ss[row * SS + col] = sc[jj][r] * 0.125f;  // 1/sqrt(64)
            }
        }
        __syncthreads();

        // ---- online softmax
        {
            const int row = tid & 63, seg = tid >> 6;
            float m = -3.0e38f;
#pragma unroll
            for (int c = 0; c < 16; ++c)
                m = fmaxf(m, Ss[row * SS + seg * 16 + c]);
            pmax[seg][row] = m;
        }
        __syncthreads();
        if (tid < 64) {
            float m = fmaxf(fmaxf(pmax[0][tid], pmax[1][tid]),
                            fmaxf(pmax[2][tid], pmax[3][tid]));
            const float mn = fmaxf(mrow[tid], m);
            arow[tid] = __expf(mrow[tid] - mn);
            mrow[tid] = mn;
        }
        __syncthreads();
        {
            const int row = tid & 63, seg = tid >> 6;
            const float mn = mrow[row];
            float s = 0.0f;
#pragma unroll
            for (int c = 0; c < 16; ++c) {
                const float p = __expf(Ss[row * SS + seg * 16 + c] - mn);
                s += p;
                Ps[row * PS + seg * 16 + c] = f2bf(p);
            }
            psum[seg][row] = s;
        }
        __syncthreads();
        if (tid < 64)
            lrow[tid] = lrow[tid] * arow[tid] +
                        (psum[0][tid] + psum[1][tid] + psum[2][tid] + psum[3][tid]);

        // rescale output accumulators by alpha
#pragma unroll
        for (int r = 0; r < 8; ++r) {
            const float al = arow[fi * 16 + r + 8 * half];
            accO[0][r] *= al;
            accO[1][r] *= al;
        }

        // ---- P(64x64) * V(64x64)
        v16bf pa0 = load_frag_a(Ps, PS, fi * 16, 0, lane);
        v16bf pa1 = load_frag_a(Ps, PS, fi * 16, 32, lane);
#pragma unroll
        for (int jj = 0; jj < 2; ++jj) {
            v16bf vb0 = load_frag_b(&Vts[cur][0], VS, (fj0 + jj) * 16, 0, lane);
            v16bf vb1 = load_frag_b(&Vts[cur][0], VS, (fj0 + jj) * 16, 32, lane);
            accO[jj] = wmma_bf16(pa0, vb0, accO[jj]);
            accO[jj] = wmma_bf16(pa1, vb1, accO[jj]);
        }
        __syncthreads();  // before next iter overwrites Ss/Ps and buf cur^1
    }

    // ---- epilogue: ctx[b][s][h*64+d] = O / l   (bf16, [B,S,D] layout)
#pragma unroll
    for (int r = 0; r < 8; ++r) {
        const int row = fi * 16 + r + 8 * half;
        const float inv = 1.0f / lrow[row];
        const int s = qt * 64 + row;
#pragma unroll
        for (int jj = 0; jj < 2; ++jj) {
            const int d = (fj0 + jj) * 16 + l15;
            ctx[((size_t)(b * SEQ + s)) * DMODEL + h * HDIM + d] =
                f2bf(accO[jj][r] * inv);
        }
    }
}

// ---------------------------------------------------------------- launch ---
extern "C" void kernel_launch(void* const* d_in, const int* in_sizes, int n_in,
                              void* d_out, int out_size, void* d_ws, size_t ws_size,
                              hipStream_t stream) {
    const float* x    = (const float*)d_in[0];
    const float* Wqkv = (const float*)d_in[1];
    const float* bqkv = (const float*)d_in[2];
    const float* Wout = (const float*)d_in[3];
    const float* bout = (const float*)d_in[4];
    float* out = (float*)d_out;

    char* ws = (char*)d_ws;
    unsigned short* xb    = (unsigned short*)(ws + XB_OFF);
    unsigned short* wqkvt = (unsigned short*)(ws + WQKVT_OFF);
    unsigned short* woutt = (unsigned short*)(ws + WOUTT_OFF);
    unsigned short* qbuf  = (unsigned short*)(ws + Q_OFF);
    unsigned short* kbuf  = (unsigned short*)(ws + K_OFF);
    unsigned short* vtbuf = (unsigned short*)(ws + VT_OFF);
    unsigned short* ctx   = (unsigned short*)(ws + CTX_OFF);

    {
        const int n = MROWS * DMODEL;
        convert_f32_bf16_kernel<<<n / (256 * 4), 256, 0, stream>>>(x, xb, n);
    }
    transpose_f32_bf16_kernel<<<dim3(THREED / 32, DMODEL / 32), dim3(32, 8), 0,
                                stream>>>(Wqkv, wqkvt, DMODEL, THREED);
    transpose_f32_bf16_kernel<<<dim3(DMODEL / 32, DMODEL / 32), dim3(32, 8), 0,
                                stream>>>(Wout, woutt, DMODEL, DMODEL);
    gemm_bf16_kernel<0><<<dim3(THREED / 128, MROWS / 128), 256, 0, stream>>>(
        xb, wqkvt, bqkv, nullptr, qbuf, kbuf, vtbuf, MROWS, THREED, DMODEL);
    attn_kernel<<<dim3(SEQ / 64, BATCH * NHEADS), 256, 0, stream>>>(qbuf, kbuf,
                                                                    vtbuf, ctx);
    gemm_bf16_kernel<1><<<dim3(DMODEL / 128, MROWS / 128), 256, 0, stream>>>(
        ctx, woutt, bout, out, nullptr, nullptr, nullptr, MROWS, DMODEL, DMODEL);
}
